// UpDownCell_52664888983687
// MI455X (gfx1250) — matrix operand order repaired and checked by reading, not compiled, not run on hardware
//
#include <hip/hip_runtime.h>
#include <hip/hip_bf16.h>
#include <math.h>

// ---------------------------------------------------------------------------
// UpDownCell (BUTD captioning cell) for MI455X / gfx1250, wave32 + WMMA.
//
// Exploits h1=c1=h2=c2=0 on the reference path:
//   * W_hh1 / W_hh2 terms are identically zero -> never read (saves 32 MB).
//   * last 2000 K-cols of W_ih1 and last 1000 of W_ih2 multiply zeros -> K=3048.
// All large GEMMs go through v_wmma_f32_16x16x32_bf16 (fp32->bf16 inline
// convert, f32 accumulate); pointwise/softmax/pool stay fp32 VALU.
//
// GEMM: one wave per 64(M)x32(N) tile, K stepped by 32. Main loop is
// branch-free (K_main = K & ~31); the ragged K tail (8 for K=3048/1000) is a
// single guarded step after the loop so the hot loop stays clean.
// ---------------------------------------------------------------------------

#define BATCH 256
#define NBOX  36
#define FIMG  2048
#define EMB   1000
#define HID   1000
#define PROJ  512
#define KEFF  3048   // EMB+FIMG for cell1 path; FIMG+HID for cell2 path

typedef __attribute__((ext_vector_type(16))) __bf16 v16bf;
typedef __attribute__((ext_vector_type(8)))  float  v8f;

// ------------------------------- WMMA GEMM ---------------------------------

__device__ __forceinline__ void cvt8(v16bf& d, int off, float4 x, float4 y) {
    d[off + 0] = (__bf16)x.x; d[off + 1] = (__bf16)x.y;
    d[off + 2] = (__bf16)x.z; d[off + 3] = (__bf16)x.w;
    d[off + 4] = (__bf16)y.x; d[off + 5] = (__bf16)y.y;
    d[off + 6] = (__bf16)y.z; d[off + 7] = (__bf16)y.w;
}

// A fragment 16x32 bf16: lanes 0-15 (row M=lane) hold K k0+0..7 in halfs 0..7
// and k0+16..23 in halfs 8..15; lanes 16-31 hold k0+8..15 / k0+24..31.
__device__ __forceinline__ v16bf load_a_fast(const float* __restrict__ Arow,
                                             int k0, int half) {
    v16bf a;
    const float4* p = (const float4*)(Arow + k0 + half * 8);
    const float4* q = (const float4*)(Arow + k0 + 16 + half * 8);
    cvt8(a, 0, p[0], p[1]);
    cvt8(a, 8, q[0], q[1]);
    return a;
}

__device__ __forceinline__ v16bf load_a_tail(const float* __restrict__ Arow,
                                             int k0, int K, int half) {
    v16bf a;
    const int kg0 = k0 + half * 8;
    const int kg1 = k0 + 16 + half * 8;
#pragma unroll
    for (int i = 0; i < 8; ++i) {
        int ka = kg0 + i, kb = kg1 + i;
        a[i]     = (__bf16)(ka < K ? Arow[ka] : 0.0f);
        a[8 + i] = (__bf16)(kb < K ? Arow[kb] : 0.0f);
    }
    return a;
}

// B fragment 32x16 bf16: lanes 0-15 (col N=lane) hold K k0+0..7 in halfs
// 0..7 and k0+16..23 in halfs 8..15; lanes 16-31 hold k0+8..15 / k0+24..31.
__device__ __forceinline__ v16bf load_b_fast(const float* __restrict__ Brow,
                                             int k0, int half) {
    v16bf b;
    const float4* p = (const float4*)(Brow + k0 + half * 8);
    const float4* q = (const float4*)(Brow + k0 + 16 + half * 8);
    cvt8(b, 0, p[0], p[1]);
    cvt8(b, 8, q[0], q[1]);
    return b;
}

__device__ __forceinline__ v16bf load_b_tail(const float* __restrict__ Brow,
                                             int k0, int K, int half) {
    v16bf b;
    const int kb = k0 + half * 8;
#pragma unroll
    for (int i = 0; i < 16; ++i) {
        int k = kb + (i < 8 ? i : i + 8);
        b[i] = (__bf16)(k < K ? Brow[k] : 0.0f);
    }
    return b;
}

// C/D 16x16 f32: lanes 0-15 col N=lane, VGPR r -> row M=r;
// lanes 16-31 col N=lane-16, VGPR r -> row M=8+r.
__device__ __forceinline__ void store_tile(float* __restrict__ C, int ldc,
                                           v8f acc, int m0, int n,
                                           const float* __restrict__ bias1,
                                           const float* __restrict__ bias2,
                                           int half) {
    float bv = (bias1 ? bias1[n] : 0.0f) + (bias2 ? bias2[n] : 0.0f);
    const int mbase = m0 + half * 8;
#pragma unroll
    for (int r = 0; r < 8; ++r)
        C[(size_t)(mbase + r) * ldc + n] = acc[r] + bv;
}

// C[M,N] = A[M,K] (row-major, lda) @ Bw[N,K]^T (row-major, ldb) + bias1 + bias2
__global__ __launch_bounds__(32)
void gemm_bf16_tn(const float* __restrict__ A, int lda,
                  const float* __restrict__ Bw, int ldb,
                  const float* __restrict__ bias1,
                  const float* __restrict__ bias2,
                  float* __restrict__ C, int ldc, int K) {
    const int lane = threadIdx.x & 31;
    const int half = lane >> 4;
    const int l16  = lane & 15;
    const int n_tile = blockIdx.x * 32;
    const int m_tile = blockIdx.y * 64;

    const float* Arow0 = A + (size_t)(m_tile + l16) * lda;
    const float* Arow1 = Arow0 + (size_t)16 * lda;
    const float* Arow2 = Arow0 + (size_t)32 * lda;
    const float* Arow3 = Arow0 + (size_t)48 * lda;
    const float* Brow0 = Bw + (size_t)(n_tile + l16) * ldb;
    const float* Brow1 = Brow0 + (size_t)16 * ldb;

    v8f acc00 = {}, acc01 = {}, acc10 = {}, acc11 = {};
    v8f acc20 = {}, acc21 = {}, acc30 = {}, acc31 = {};

    const int K_main = K & ~31;
    for (int k0 = 0; k0 < K_main; k0 += 32) {
        v16bf b0 = load_b_fast(Brow0, k0, half);
        v16bf b1 = load_b_fast(Brow1, k0, half);
        v16bf a0 = load_a_fast(Arow0, k0, half);
        v16bf a1 = load_a_fast(Arow1, k0, half);
        v16bf a2 = load_a_fast(Arow2, k0, half);
        v16bf a3 = load_a_fast(Arow3, k0, half);
        acc00 = __builtin_amdgcn_wmma_f32_16x16x32_bf16(false, a0, false, b0,
                                                        (short)0, acc00, false, false);
        acc01 = __builtin_amdgcn_wmma_f32_16x16x32_bf16(false, a0, false, b1,
                                                        (short)0, acc01, false, false);
        acc10 = __builtin_amdgcn_wmma_f32_16x16x32_bf16(false, a1, false, b0,
                                                        (short)0, acc10, false, false);
        acc11 = __builtin_amdgcn_wmma_f32_16x16x32_bf16(false, a1, false, b1,
                                                        (short)0, acc11, false, false);
        acc20 = __builtin_amdgcn_wmma_f32_16x16x32_bf16(false, a2, false, b0,
                                                        (short)0, acc20, false, false);
        acc21 = __builtin_amdgcn_wmma_f32_16x16x32_bf16(false, a2, false, b1,
                                                        (short)0, acc21, false, false);
        acc30 = __builtin_amdgcn_wmma_f32_16x16x32_bf16(false, a3, false, b0,
                                                        (short)0, acc30, false, false);
        acc31 = __builtin_amdgcn_wmma_f32_16x16x32_bf16(false, a3, false, b1,
                                                        (short)0, acc31, false, false);
    }
    if (K_main < K) {  // single guarded tail step (K % 32 != 0)
        const int k0 = K_main;
        v16bf b0 = load_b_tail(Brow0, k0, K, half);
        v16bf b1 = load_b_tail(Brow1, k0, K, half);
        v16bf a0 = load_a_tail(Arow0, k0, K, half);
        v16bf a1 = load_a_tail(Arow1, k0, K, half);
        v16bf a2 = load_a_tail(Arow2, k0, K, half);
        v16bf a3 = load_a_tail(Arow3, k0, K, half);
        acc00 = __builtin_amdgcn_wmma_f32_16x16x32_bf16(false, a0, false, b0,
                                                        (short)0, acc00, false, false);
        acc01 = __builtin_amdgcn_wmma_f32_16x16x32_bf16(false, a0, false, b1,
                                                        (short)0, acc01, false, false);
        acc10 = __builtin_amdgcn_wmma_f32_16x16x32_bf16(false, a1, false, b0,
                                                        (short)0, acc10, false, false);
        acc11 = __builtin_amdgcn_wmma_f32_16x16x32_bf16(false, a1, false, b1,
                                                        (short)0, acc11, false, false);
        acc20 = __builtin_amdgcn_wmma_f32_16x16x32_bf16(false, a2, false, b0,
                                                        (short)0, acc20, false, false);
        acc21 = __builtin_amdgcn_wmma_f32_16x16x32_bf16(false, a2, false, b1,
                                                        (short)0, acc21, false, false);
        acc30 = __builtin_amdgcn_wmma_f32_16x16x32_bf16(false, a3, false, b0,
                                                        (short)0, acc30, false, false);
        acc31 = __builtin_amdgcn_wmma_f32_16x16x32_bf16(false, a3, false, b1,
                                                        (short)0, acc31, false, false);
    }
    store_tile(C, ldc, acc00, m_tile,      n_tile + l16,      bias1, bias2, half);
    store_tile(C, ldc, acc01, m_tile,      n_tile + 16 + l16, bias1, bias2, half);
    store_tile(C, ldc, acc10, m_tile + 16, n_tile + l16,      bias1, bias2, half);
    store_tile(C, ldc, acc11, m_tile + 16, n_tile + 16 + l16, bias1, bias2, half);
    store_tile(C, ldc, acc20, m_tile + 32, n_tile + l16,      bias1, bias2, half);
    store_tile(C, ldc, acc21, m_tile + 32, n_tile + 16 + l16, bias1, bias2, half);
    store_tile(C, ldc, acc30, m_tile + 48, n_tile + l16,      bias1, bias2, half);
    store_tile(C, ldc, acc31, m_tile + 48, n_tile + 16 + l16, bias1, bias2, half);
}

// ----------------------------- pointwise stages -----------------------------

// mask[b,n] = (sum_f |x[b,n,f]|) > 0 ? 1 : 0
__global__ __launch_bounds__(256)
void mask_kernel(const float* __restrict__ img, float* __restrict__ maskf) {
    __shared__ float red[256];
    const int bn = blockIdx.x;
    const float* p = img + (size_t)bn * FIMG;
    float s = 0.0f;
    for (int i = threadIdx.x; i < FIMG; i += 256) s += fabsf(p[i]);
    red[threadIdx.x] = s;
    __syncthreads();
    for (int o = 128; o > 0; o >>= 1) {
        if (threadIdx.x < o) red[threadIdx.x] += red[threadIdx.x + o];
        __syncthreads();
    }
    if (threadIdx.x == 0) maskf[bn] = red[0] > 0.0f ? 1.0f : 0.0f;
}

// avg[b,f] = sum_n mask[b,n]*x[b,n,f] / max(sum_n mask[b,n], 1e-13)
__global__ __launch_bounds__(256)
void avg_kernel(const float* __restrict__ img, const float* __restrict__ maskf,
                float* __restrict__ avg) {
    __shared__ float m[NBOX];
    const int b = blockIdx.x;
    if (threadIdx.x < NBOX) m[threadIdx.x] = maskf[b * NBOX + threadIdx.x];
    __syncthreads();
    float denom = 0.0f;
#pragma unroll
    for (int n = 0; n < NBOX; ++n) denom += m[n];
    const float inv = 1.0f / fmaxf(denom, 1e-13f);
    for (int f = threadIdx.x; f < FIMG; f += 256) {
        float s = 0.0f;
#pragma unroll 4
        for (int n = 0; n < NBOX; ++n)
            s += m[n] * img[((size_t)b * NBOX + n) * FIMG + f];
        avg[(size_t)b * FIMG + f] = s * inv;
    }
}

// Xa[b] = [token_embedding[b] (E), avg_feats[b] (F)]   (h1,h2 parts are 0)
__global__ __launch_bounds__(256)
void concat1_kernel(const float* __restrict__ tok, const float* __restrict__ avg,
                    float* __restrict__ Xa) {
    const int b = blockIdx.x;
    float* xr = Xa + (size_t)b * KEFF;
    for (int i = threadIdx.x; i < EMB;  i += 256) xr[i]       = tok[(size_t)b * EMB + i];
    for (int i = threadIdx.x; i < FIMG; i += 256) xr[EMB + i] = avg[(size_t)b * FIMG + i];
}

// X2[b] = [attended[b] (F), h1[b] (H)]   (h2 part is 0)
__global__ __launch_bounds__(256)
void concat2_kernel(const float* __restrict__ att, const float* __restrict__ h1,
                    float* __restrict__ X2) {
    const int b = blockIdx.x;
    float* xr = X2 + (size_t)b * KEFF;
    for (int i = threadIdx.x; i < FIMG; i += 256) xr[i]        = att[(size_t)b * FIMG + i];
    for (int i = threadIdx.x; i < HID;  i += 256) xr[FIMG + i] = h1[(size_t)b * HID + i];
}

// LSTM cell with c_prev=0: h = sigmoid(o) * tanh(sigmoid(i) * tanh(g))
__global__ __launch_bounds__(256)
void lstm_act_kernel(const float* __restrict__ gates, float* __restrict__ h) {
    const int idx = blockIdx.x * 256 + threadIdx.x;
    if (idx >= BATCH * HID) return;
    const int b = idx / HID, j = idx - b * HID;
    const float* g = gates + (size_t)b * 4 * HID;
    const float gi = g[j];
    const float gg = g[2 * HID + j];
    const float go = g[3 * HID + j];
    const float si = 1.0f / (1.0f + expf(-gi));
    const float c  = si * tanhf(gg);
    const float so = 1.0f / (1.0f + expf(-go));
    h[idx] = so * tanhf(c);
}

// scores[b,n] = sum_p tanh(q[b,p]+v[b,n,p])*wa[p]; masked softmax over n
__global__ __launch_bounds__(256)
void attn_scores_kernel(const float* __restrict__ qproj,
                        const float* __restrict__ vproj,
                        const float* __restrict__ wa,
                        const float* __restrict__ maskf,
                        float* __restrict__ attn) {
    __shared__ float q[PROJ], w[PROJ], sc[NBOX];
    const int b = blockIdx.x;
    for (int i = threadIdx.x; i < PROJ; i += 256) {
        q[i] = qproj[(size_t)b * PROJ + i];
        w[i] = wa[i];
    }
    __syncthreads();
    const int wv = threadIdx.x >> 5, lane = threadIdx.x & 31;
    for (int n = wv; n < NBOX; n += 8) {
        const float* v = vproj + ((size_t)b * NBOX + n) * PROJ;
        float s = 0.0f;
        for (int p = lane; p < PROJ; p += 32) s += tanhf(q[p] + v[p]) * w[p];
        for (int o = 16; o > 0; o >>= 1) s += __shfl_down(s, o, 32);
        if (lane == 0)
            sc[n] = (maskf[b * NBOX + n] > 0.5f) ? s : -1e9f;
    }
    __syncthreads();
    if (threadIdx.x == 0) {
        float mx = -1e30f;
        for (int n = 0; n < NBOX; ++n) mx = fmaxf(mx, sc[n]);
        float sum = 0.0f;
        for (int n = 0; n < NBOX; ++n) { float e = expf(sc[n] - mx); sc[n] = e; sum += e; }
        const float inv = 1.0f / sum;
        for (int n = 0; n < NBOX; ++n) attn[b * NBOX + n] = sc[n] * inv;
    }
}

// attended[b,f] = sum_n attn[b,n] * img[b,n,f]
__global__ __launch_bounds__(256)
void attended_kernel(const float* __restrict__ attn, const float* __restrict__ img,
                     float* __restrict__ out) {
    __shared__ float a[NBOX];
    const int b = blockIdx.x;
    if (threadIdx.x < NBOX) a[threadIdx.x] = attn[b * NBOX + threadIdx.x];
    __syncthreads();
    for (int f = threadIdx.x; f < FIMG; f += 256) {
        float s = 0.0f;
#pragma unroll 4
        for (int n = 0; n < NBOX; ++n)
            s += a[n] * img[((size_t)b * NBOX + n) * FIMG + f];
        out[(size_t)b * FIMG + f] = s;
    }
}

// ------------------------------- host launch --------------------------------

extern "C" void kernel_launch(void* const* d_in, const int* in_sizes, int n_in,
                              void* d_out, int out_size, void* d_ws, size_t ws_size,
                              hipStream_t stream) {
    const float* img   = (const float*)d_in[0];   // [B,NB,F]
    const float* tok   = (const float*)d_in[1];   // [B,E]
    const float* W_ih1 = (const float*)d_in[2];   // [4H, 5048]
    const float* b_ih1 = (const float*)d_in[3];   // [4H]
    /* d_in[4] = W_hh1 : unused (h1_prev == 0) */
    const float* b_hh1 = (const float*)d_in[5];   // [4H]
    const float* Wq    = (const float*)d_in[6];   // [P,H]
    const float* bq    = (const float*)d_in[7];   // [P]
    const float* Wv    = (const float*)d_in[8];   // [P,F]
    const float* bv    = (const float*)d_in[9];   // [P]
    const float* wa    = (const float*)d_in[10];  // [P]
    const float* W_ih2 = (const float*)d_in[11];  // [4H, 4048]
    const float* b_ih2 = (const float*)d_in[12];  // [4H]
    /* d_in[13] = W_hh2 : unused (h2_prev == 0) */
    const float* b_hh2 = (const float*)d_in[14];  // [4H]
    float* out = (float*)d_out;                   // [B,H]
    (void)in_sizes; (void)n_in; (void)out_size; (void)ws_size;

    char* ws = (char*)d_ws;
    size_t off = 0;
    auto alloc = [&](size_t bytes) -> float* {
        off = (off + 255) & ~(size_t)255;
        float* p = (float*)(ws + off);
        off += bytes;
        return p;
    };
    float* maskf = alloc((size_t)BATCH * NBOX * 4);
    float* avg   = alloc((size_t)BATCH * FIMG * 4);
    float* Xa    = alloc((size_t)BATCH * KEFF * 4);    // reused for X2
    float* gates = alloc((size_t)BATCH * 4 * HID * 4); // reused for both cells
    float* h1    = alloc((size_t)BATCH * HID * 4);
    float* qp    = alloc((size_t)BATCH * PROJ * 4);
    float* vp    = alloc((size_t)BATCH * NBOX * PROJ * 4);
    float* attn  = alloc((size_t)BATCH * NBOX * 4);
    float* att   = alloc((size_t)BATCH * FIMG * 4);

    // 1) masked mean pool
    mask_kernel<<<BATCH * NBOX, 256, 0, stream>>>(img, maskf);
    avg_kernel<<<BATCH, 256, 0, stream>>>(img, maskf, avg);

    // 2) attention LSTM: gates1 = Xa @ W_ih1[:, :3048]^T + b_ih1 + b_hh1
    concat1_kernel<<<BATCH, 256, 0, stream>>>(tok, avg, Xa);
    gemm_bf16_tn<<<dim3(4 * HID / 32, BATCH / 64), 32, 0, stream>>>(
        Xa, KEFF, W_ih1, EMB + FIMG + 2 * HID, b_ih1, b_hh1, gates, 4 * HID, KEFF);
    lstm_act_kernel<<<(BATCH * HID + 255) / 256, 256, 0, stream>>>(gates, h1);

    // 3) BUTD attention
    gemm_bf16_tn<<<dim3(PROJ / 32, BATCH / 64), 32, 0, stream>>>(
        h1, HID, Wq, HID, bq, nullptr, qp, PROJ, HID);
    gemm_bf16_tn<<<dim3(PROJ / 32, (BATCH * NBOX) / 64), 32, 0, stream>>>(
        img, FIMG, Wv, FIMG, bv, nullptr, vp, PROJ, FIMG);
    attn_scores_kernel<<<BATCH, 256, 0, stream>>>(qp, vp, wa, maskf, attn);
    attended_kernel<<<BATCH, 256, 0, stream>>>(attn, img, att);

    // 4) language LSTM: gates2 = X2 @ W_ih2[:, :3048]^T + b_ih2 + b_hh2
    concat2_kernel<<<BATCH, 256, 0, stream>>>(att, h1, Xa);
    gemm_bf16_tn<<<dim3(4 * HID / 32, BATCH / 64), 32, 0, stream>>>(
        Xa, KEFF, W_ih2, FIMG + 2 * HID, b_ih2, b_hh2, gates, 4 * HID, KEFF);
    lstm_act_kernel<<<(BATCH * HID + 255) / 256, 256, 0, stream>>>(gates, out);
}
